// TextualGraph_59605556134144
// MI455X (gfx1250) — matrix operand order repaired and checked by reading, not compiled, not run on hardware
//
#include <hip/hip_runtime.h>
#include <hip/hip_bf16.h>
#include <math.h>

// ---------------- CDNA5 WMMA types ----------------
typedef __bf16 bf16;
typedef __attribute__((ext_vector_type(2)))  __bf16 v2bf;
typedef __attribute__((ext_vector_type(16))) __bf16 v16bf;
typedef __attribute__((ext_vector_type(8)))  float  v8f;

#define WMMA_BF16(A, B, C) \
  __builtin_amdgcn_wmma_f32_16x16x32_bf16(false, (A), false, (B), (short)0, (C), false, false)

// A-fragment K offset for 16-bit A 16x32 layout (matches B):
// lanes 0-15: VGPR v -> K=2v+e ; lanes 16-31: +16
__device__ __forceinline__ int a_koff(int v, int half) {
  return (half << 4) + (v << 1);
}
// B-fragment: lane n = lane&15 is the column; lanes 0-15 K=0..15, 16-31 K=16..31,
// VGPR v packs K = kb0 + (half<<4) + 2v + e.
// D tile: lane col = lane&15, row = v + 8*(lane>=16).

// ---------------- dims ----------------
#define B_IMG 64
#define R_REG 36
#define E_DIM 1024
#define C_CAP 64
#define L_W   32
#define N_DEP 30
#define NB    32
#define HID   512

// =====================================================================
// K0: convert the three 512x512 weights to bf16 (kept row-major (n,k))
// =====================================================================
__global__ __launch_bounds__(256) void k_convert_w(
    const float* __restrict__ W1, const float* __restrict__ Wmu,
    const float* __restrict__ Wlv,
    bf16* __restrict__ o1, bf16* __restrict__ omu, bf16* __restrict__ olv) {
  int i = blockIdx.x * 256 + threadIdx.x;
  if (i < HID * HID) {
    o1[i]  = (bf16)W1[i];
    omu[i] = (bf16)Wmu[i];
    olv[i] = (bf16)Wlv[i];
  }
}

// =====================================================================
// K1: per (caption c, image b): attention scores -> leaky -> l2norm(words)
//     -> softmax(regions) -> weighted context -> blockwise cosine tnode
// grid 4096 blocks, 256 threads (8 waves)
// =====================================================================
__global__ __launch_bounds__(256) void k_attn_tnode(
    const float* __restrict__ images, const float* __restrict__ captions,
    float* __restrict__ tnode) {
  __shared__ float S[48][32];   // padded region-word scores (rows >=36 are 0)
  __shared__ bf16  A2[32][64];  // softmaxed attention, K padded 36->64

  int cb = blockIdx.x;
  int c = cb >> 6, b = cb & 63;
  const float* imgB = images   + (size_t)b * R_REG * E_DIM;
  const float* capC = captions + (size_t)c * L_W * E_DIM;

  int lane = threadIdx.x & 31;
  int wv   = threadIdx.x >> 5;
  int half = lane >> 4;
  int lrow = lane & 15;

  // ---- Phase 1: S = images[b] @ cap[c]^T  (36x32, padded 48x32), bf16 WMMA
  // OOB rows: clamp address (legal load) + multiply by 0/1 mask -> no EXEC
  // divergence, loads stay a flat pipelined stream.
  if (wv < 6) {
    int m0 = (wv >> 1) << 4;  // 0,16,32
    int n0 = (wv & 1) << 4;   // 0,16
    int am = m0 + lrow;
    float amask = (am < R_REG) ? 1.f : 0.f;
    const float* arow = imgB + (size_t)((am < R_REG) ? am : 0) * E_DIM;
    const float* brow = capC + (size_t)(n0 + lrow) * E_DIM;
    v8f acc = {};
    for (int ks = 0; ks < 32; ++ks) {
      int kb0 = ks << 5;
      v16bf af, bfr;
#pragma unroll
      for (int v = 0; v < 8; ++v) {
        int ka = kb0 + a_koff(v, half);
        float2 a2 = *(const float2*)(arow + ka);
        af[2 * v] = (bf16)(a2.x * amask); af[2 * v + 1] = (bf16)(a2.y * amask);
        int kb = kb0 + (half << 4) + (v << 1);
        float2 b2v = *(const float2*)(brow + kb);
        bfr[2 * v] = (bf16)b2v.x; bfr[2 * v + 1] = (bf16)b2v.y;
      }
      acc = WMMA_BF16(af, bfr, acc);
    }
#pragma unroll
    for (int v = 0; v < 8; ++v)
      S[m0 + v + (half << 3)][n0 + lrow] = acc[v];
  }
  __syncthreads();

  // ---- Phase 2: LeakyReLU(0.1) then l2norm over words (per region row)
  if (threadIdx.x < R_REG) {
    int r = threadIdx.x;
    float vals[32];
    float ss = 0.f;
#pragma unroll
    for (int l = 0; l < 32; ++l) {
      float x = S[r][l];
      x = (x >= 0.f) ? x : 0.1f * x;
      vals[l] = x; ss += x * x;
    }
    float inv = 1.f / (sqrtf(ss) + 1e-8f);
#pragma unroll
    for (int l = 0; l < 32; ++l) S[r][l] = vals[l] * inv;
  }
  __syncthreads();

  // ---- Phase 3: softmax over regions per word, write A2[l][r] bf16 (pad->0)
  if (threadIdx.x < L_W) {
    int l = threadIdx.x;
    float mx = -3.4e38f;
#pragma unroll
    for (int r = 0; r < R_REG; ++r) mx = fmaxf(mx, 9.f * S[r][l]);
    float ex[R_REG]; float sum = 0.f;
#pragma unroll
    for (int r = 0; r < R_REG; ++r) { float e = __expf(9.f * S[r][l] - mx); ex[r] = e; sum += e; }
    float inv = 1.f / sum;
#pragma unroll
    for (int r = 0; r < R_REG; ++r) A2[l][r] = (bf16)(ex[r] * inv);
#pragma unroll
    for (int r = R_REG; r < 64; ++r) A2[l][r] = (bf16)0.f;
  }
  __syncthreads();

  // ---- Phase 4: wctx = A2 @ images[b]  with fused blockwise-cosine tnode.
  // wave wv owns columns [wv*128, wv*128+128) = 4 blocks of 32; 2 M-tiles.
#pragma unroll 1
  for (int mt = 0; mt < 2; ++mt) {
    v16bf a0, a1;
    int ar = (mt << 4) + lrow;
#pragma unroll
    for (int v = 0; v < 8; ++v) {
      int ka = a_koff(v, half);
      a0[2 * v] = A2[ar][ka];      a0[2 * v + 1] = A2[ar][ka + 1];
      a1[2 * v] = A2[ar][ka + 32]; a1[2 * v + 1] = A2[ar][ka + 33];
    }
    v8f dprev = {};
#pragma unroll 1
    for (int nt = 0; nt < 8; ++nt) {
      int n0 = (wv << 7) + (nt << 4);
      int bcol = n0 + lrow;
      v16bf b0, b1;
#pragma unroll
      for (int v = 0; v < 8; ++v) {
        int k0 = (half << 4) + (v << 1);  // 0..31 -> always < 36
        b0[2 * v]     = (bf16)imgB[(size_t)k0 * E_DIM + bcol];
        b0[2 * v + 1] = (bf16)imgB[(size_t)(k0 + 1) * E_DIM + bcol];
        // K 32..63: clamp address, zero value via cndmask (no branches)
        int k1 = k0 + 32;
        int k1c0 = (k1 < R_REG) ? k1 : (R_REG - 1);
        int k1c1 = (k1 + 1 < R_REG) ? (k1 + 1) : (R_REG - 1);
        float x0 = imgB[(size_t)k1c0 * E_DIM + bcol];
        float x1 = imgB[(size_t)k1c1 * E_DIM + bcol];
        b1[2 * v]     = (bf16)((k1 < R_REG) ? x0 : 0.f);
        b1[2 * v + 1] = (bf16)((k1 + 1 < R_REG) ? x1 : 0.f);
      }
      v8f d = {};
      d = WMMA_BF16(a0, b0, d);
      d = WMMA_BF16(a1, b1, d);
      if ((nt & 1) == 0) {
        dprev = d;
      } else {
        // block f covers columns [fbase, fbase+32): tiles dprev(+0..15) & d(+16..31)
        int fbase = (wv << 7) + ((nt - 1) << 4);
#pragma unroll
        for (int v = 0; v < 8; ++v) {
          int l = (mt << 4) + v + (half << 3);
          float q0 = capC[(size_t)l * E_DIM + fbase + lrow];
          float q1 = capC[(size_t)l * E_DIM + fbase + 16 + lrow];
          float c0 = dprev[v], c1 = d[v];
          float num = c0 * q0 + c1 * q1;
          float cc  = c0 * c0 + c1 * c1;
          float qq  = q0 * q0 + q1 * q1;
#pragma unroll
          for (int off = 8; off >= 1; off >>= 1) {
            num += __shfl_xor(num, off, 16);
            cc  += __shfl_xor(cc,  off, 16);
            qq  += __shfl_xor(qq,  off, 16);
          }
          if (lrow == 0) {
            tnode[((((size_t)c * B_IMG + b) * L_W + l) << 5) + (fbase >> 5)] =
                num / fmaxf(sqrtf(cc) * sqrtf(qq), 1e-8f);
          }
        }
      }
    }
  }
}

// =====================================================================
// K2: per caption: words_sim softmax, sparse adjacency, s[l] scalars
// =====================================================================
__global__ __launch_bounds__(256) void k_graph_s(
    const float* __restrict__ captions, const int* __restrict__ depends,
    float* __restrict__ s_ws) {
  __shared__ float G[32][32];
  __shared__ float Adj[32][32];
  int c = blockIdx.x;
  const float* capC = captions + (size_t)c * L_W * E_DIM;
  int t = threadIdx.x;

#pragma unroll
  for (int e = 0; e < 4; ++e) ((float*)Adj)[t * 4 + e] = 0.f;

#pragma unroll 1
  for (int e = 0; e < 4; ++e) {
    int idx = t * 4 + e;
    int i = idx >> 5, j = idx & 31;
    const float4* ci = (const float4*)(capC + (size_t)i * E_DIM);
    const float4* cj = (const float4*)(capC + (size_t)j * E_DIM);
    float s = 0.f;
    for (int q = 0; q < E_DIM / 4; ++q) {
      float4 a = ci[q], b4 = cj[q];
      s += a.x * b4.x + a.y * b4.y + a.z * b4.z + a.w * b4.w;
    }
    G[i][j] = s;
  }
  __syncthreads();

  if (t > 0 && t < N_DEP) {  // dep_mask skips pair 0
    int d0 = depends[((size_t)c * N_DEP + t) * 2 + 0];
    int d1 = depends[((size_t)c * N_DEP + t) * 2 + 1];
    Adj[d0][d1] = 1.f;
    Adj[d1][d0] = 1.f;
  }
  __syncthreads();
  if (t < 32) Adj[t][t] += 1.f;  // + eye (after scatter-max)
  __syncthreads();

  if (t < 32) {  // row softmax of 9*G in place
    float mx = -3.4e38f;
#pragma unroll
    for (int j = 0; j < 32; ++j) mx = fmaxf(mx, 9.f * G[t][j]);
    float sum = 0.f;
#pragma unroll
    for (int j = 0; j < 32; ++j) { float e = __expf(9.f * G[t][j] - mx); G[t][j] = e; sum += e; }
    float inv = 1.f / sum;
#pragma unroll
    for (int j = 0; j < 32; ++j) G[t][j] *= inv;
  }
  __syncthreads();

  if (t < 32) {
    float num = 0.f, ss = 0.f;
#pragma unroll
    for (int n = 0; n < 32; ++n) {
      float m = Adj[t][n] * G[t][n];
      num += m * Adj[t][n];
      ss  += m * m;
    }
    s_ws[(c << 5) + t] = num / (sqrtf(ss) + 1e-8f);
  }
}

// =====================================================================
// K3: hidden[c] = (tnode[c]*s[c]) @ W_gc^T  -> bf16.  One wave per 16x16 tile.
// 262144 waves = 32768 blocks x 8 waves.
// =====================================================================
__global__ __launch_bounds__(256) void k_hidden(
    const float* __restrict__ tnode, const float* __restrict__ s_ws,
    const float* __restrict__ W_gc, bf16* __restrict__ hidden) {
  int lane = threadIdx.x & 31;
  int wv   = threadIdx.x >> 5;
  int gw   = blockIdx.x * 8 + wv;   // c(6b) | mtile(7b) | ntile(5b)
  int c    = gw >> 12;
  int rem  = gw & 4095;
  int m0   = (rem >> 5) << 4;
  int n0   = (rem & 31) << 4;
  int half = lane >> 4, lrow = lane & 15;

  int m = m0 + lrow;                // m = b*32 + l
  int bi = m >> 5, l = m & 31;
  float sc = s_ws[(c << 5) + l];
  const float* tn = tnode + ((((size_t)c * B_IMG + bi) * L_W + l) << 5);

  v16bf af, bfr;
#pragma unroll
  for (int v = 0; v < 8; ++v) {
    int ka = a_koff(v, half);       // K = 0..31, single step
    float2 t2 = *(const float2*)(tn + ka);
    af[2 * v] = (bf16)(t2.x * sc); af[2 * v + 1] = (bf16)(t2.y * sc);
  }
  const float* wg = W_gc + (size_t)(n0 + lrow) * NB;
#pragma unroll
  for (int v = 0; v < 8; ++v) {
    int kb = (half << 4) + (v << 1);
    float2 w2 = *(const float2*)(wg + kb);
    bfr[2 * v] = (bf16)w2.x; bfr[2 * v + 1] = (bf16)w2.y;
  }
  v8f d = {};
  d = WMMA_BF16(af, bfr, d);

  bf16* hd = hidden + ((size_t)c << 11) * HID;
#pragma unroll
  for (int v = 0; v < 8; ++v)
    hd[(size_t)(m0 + v + (half << 3)) * HID + n0 + lrow] = (bf16)d[v];
}

// =====================================================================
// K4: rowsum[c,m] = sum_n tanh(hidden@W1^T + b1)[m,n] * W2[n]   (h2 fused away)
// One wave per (c, mtile): 8192 waves = 1024 blocks x 8 waves.
// =====================================================================
__global__ __launch_bounds__(256) void k_simrow(
    const bf16* __restrict__ hidden, const bf16* __restrict__ Wb1,
    const float* __restrict__ b1, const float* __restrict__ W2,
    float* __restrict__ rowsum) {
  int lane = threadIdx.x & 31;
  int wv   = threadIdx.x >> 5;
  int gw   = blockIdx.x * 8 + wv;
  int c    = gw >> 7;
  int m0   = (gw & 127) << 4;
  int half = lane >> 4, lrow = lane & 15;

  const bf16* ha = hidden + ((size_t)c << 11) * HID + (size_t)(m0 + lrow) * HID;
  v8f racc = {};
#pragma unroll 1
  for (int nt = 0; nt < 32; ++nt) {
    int n0 = nt << 4;
    const bf16* wb = Wb1 + (size_t)(n0 + lrow) * HID;
    v8f d = {};
#pragma unroll 1
    for (int ks = 0; ks < 32; ++ks) {
      int kb0 = ks << 5;
      v16bf af, bfr;
#pragma unroll
      for (int v = 0; v < 8; ++v) {
        int ka = kb0 + a_koff(v, half);
        v2bf a2 = *(const v2bf*)(ha + ka);
        af[2 * v] = a2[0]; af[2 * v + 1] = a2[1];
        int kb = kb0 + (half << 4) + (v << 1);
        v2bf b2v = *(const v2bf*)(wb + kb);
        bfr[2 * v] = b2v[0]; bfr[2 * v + 1] = b2v[1];
      }
      d = WMMA_BF16(af, bfr, d);
    }
    float b1v = b1[n0 + lrow];
    float w2v = W2[n0 + lrow];
#pragma unroll
    for (int v = 0; v < 8; ++v)
      racc[v] += tanhf(d[v] + b1v) * w2v;
  }
#pragma unroll
  for (int v = 0; v < 8; ++v) {
    float x = racc[v];
#pragma unroll
    for (int off = 8; off >= 1; off >>= 1) x += __shfl_xor(x, off, 16);
    if (lrow == 0)
      rowsum[(c << 11) + m0 + v + (half << 3)] = x;
  }
}

// =====================================================================
// K5: similarities (output slice 0) + softmax weights w[b,c]
// =====================================================================
__global__ __launch_bounds__(64) void k_sims(
    const float* __restrict__ rowsum, const float* __restrict__ b2,
    float* __restrict__ out_sim, float* __restrict__ w_ws) {
  __shared__ float sh[64];
  int b = blockIdx.x, c = threadIdx.x;
  float sum = 0.f;
#pragma unroll
  for (int l = 0; l < 32; ++l) sum += rowsum[(c << 11) + (b << 5) + l];
  float sim = b2[0] + sum * (1.f / 32.f);
  out_sim[(b << 6) + c] = sim;
  sh[c] = sim;
  __syncthreads();
  float mx = -3.4e38f;
  for (int i = 0; i < 64; ++i) mx = fmaxf(mx, sh[i]);
  float e = __expf(sim - mx);
  __syncthreads();
  sh[c] = e;
  __syncthreads();
  float se = 0.f;
  for (int i = 0; i < 64; ++i) se += sh[i];
  w_ws[(b << 6) + c] = e / se;
}

// =====================================================================
// K6: mu_word / sigma_word. One wave per output 16x16 tile; caption loop
// innermost with register accumulation (dual WMMA sharing A-fragment).
// 4096 waves = 512 blocks x 8 waves.
// =====================================================================
__global__ __launch_bounds__(256) void k_musig(
    const bf16* __restrict__ hidden, const bf16* __restrict__ Wbmu,
    const bf16* __restrict__ Wblv, const float* __restrict__ b_mu,
    const float* __restrict__ b_lv, const float* __restrict__ w_ws,
    float* __restrict__ out) {
  int lane = threadIdx.x & 31;
  int wv   = threadIdx.x >> 5;
  int gw   = blockIdx.x * 8 + wv;   // mtile(7b) | ntile(5b)
  int m0   = (gw >> 5) << 4;
  int n0   = (gw & 31) << 4;
  int half = lane >> 4, lrow = lane & 15;
  int bi   = m0 >> 5;               // tile never crosses an image boundary

  float bmu = b_mu[n0 + lrow], blv = b_lv[n0 + lrow];
  const bf16* wmu = Wbmu + (size_t)(n0 + lrow) * HID;
  const bf16* wlv = Wblv + (size_t)(n0 + lrow) * HID;

  v8f amu = {}, am2 = {};
#pragma unroll 1
  for (int c = 0; c < C_CAP; ++c) {
    float wgt = w_ws[(bi << 6) + c];
    const bf16* ha = hidden + ((size_t)c << 11) * HID + (size_t)(m0 + lrow) * HID;
    v8f dmu = {}, dlv = {};
#pragma unroll 1
    for (int ks = 0; ks < 32; ++ks) {
      int kb0 = ks << 5;
      v16bf af, bmf, blf;
#pragma unroll
      for (int v = 0; v < 8; ++v) {
        int ka = kb0 + a_koff(v, half);
        v2bf a2 = *(const v2bf*)(ha + ka);
        af[2 * v] = a2[0]; af[2 * v + 1] = a2[1];
        int kb = kb0 + (half << 4) + (v << 1);
        v2bf m2 = *(const v2bf*)(wmu + kb);
        bmf[2 * v] = m2[0]; bmf[2 * v + 1] = m2[1];
        v2bf l2 = *(const v2bf*)(wlv + kb);
        blf[2 * v] = l2[0]; blf[2 * v + 1] = l2[1];
      }
      dmu = WMMA_BF16(af, bmf, dmu);
      dlv = WMMA_BF16(af, blf, dlv);
    }
#pragma unroll
    for (int v = 0; v < 8; ++v) {
      float muv = dmu[v] + bmu;
      float s2  = __expf(dlv[v] + blv);   // sigma^2 = exp(logvar)
      amu[v] += wgt * muv;
      am2[v] += wgt * (s2 + muv * muv);
    }
  }
#pragma unroll
  for (int v = 0; v < 8; ++v) {
    int m = m0 + v + (half << 3);
    size_t idx = (size_t)m * HID + n0 + lrow;
    float mu = amu[v];
    float var = am2[v] - mu * mu;
    out[4096 + idx] = mu;                                   // mu_word
    out[4096 + 1048576 + idx] = sqrtf(fmaxf(var, 1e-8f));   // sigma_word
  }
}

// =====================================================================
extern "C" void kernel_launch(void* const* d_in, const int* in_sizes, int n_in,
                              void* d_out, int out_size, void* d_ws, size_t ws_size,
                              hipStream_t stream) {
  (void)in_sizes; (void)n_in; (void)out_size; (void)ws_size;
  const float* images   = (const float*)d_in[0];
  const float* captions = (const float*)d_in[1];
  const int*   depends  = (const int*)d_in[2];
  const float* W_gc     = (const float*)d_in[3];
  const float* W_mu     = (const float*)d_in[4];
  const float* b_mu     = (const float*)d_in[5];
  const float* W_lv     = (const float*)d_in[6];
  const float* b_lv     = (const float*)d_in[7];
  const float* W1       = (const float*)d_in[8];
  const float* b1       = (const float*)d_in[9];
  const float* W2       = (const float*)d_in[10];
  const float* b2       = (const float*)d_in[11];
  float* out = (float*)d_out;

  char* ws = (char*)d_ws;
  bf16*  hidden = (bf16*)(ws);                      // 64*2048*512 bf16 = 128 MiB
  float* tnode  = (float*)(ws + 134217728);         // 64*64*32*32 f32 = 16 MiB
  bf16*  Wb1    = (bf16*)(ws + 150994944);          // 512 KiB
  bf16*  Wbmu   = (bf16*)(ws + 151519232);          // 512 KiB
  bf16*  Wblv   = (bf16*)(ws + 152043520);          // 512 KiB
  float* s_ws   = (float*)(ws + 152567808);         // 8 KiB
  float* rowsum = (float*)(ws + 152576000);         // 512 KiB
  float* w_ws   = (float*)(ws + 153100288);         // 16 KiB

  k_convert_w<<<(HID * HID + 255) / 256, 256, 0, stream>>>(W1, W_mu, W_lv, Wb1, Wbmu, Wblv);
  k_attn_tnode<<<C_CAP * B_IMG, 256, 0, stream>>>(images, captions, tnode);
  k_graph_s<<<C_CAP, 256, 0, stream>>>(captions, depends, s_ws);
  k_hidden<<<32768, 256, 0, stream>>>(tnode, s_ws, W_gc, hidden);
  k_simrow<<<1024, 256, 0, stream>>>(hidden, Wb1, b1, W2, rowsum);
  k_sims<<<B_IMG, 64, 0, stream>>>(rowsum, b2, out, w_ws);
  k_musig<<<512, 256, 0, stream>>>(hidden, Wbmu, Wblv, b_mu, b_lv, w_ws, out);
}
